// LocalInferenceLayer_24240795419048
// MI455X (gfx1250) — matrix operand order repaired and checked by reading, not compiled, not run on hardware
//
#include <hip/hip_runtime.h>

// Problem constants (from reference): B=16, LA=LB=1024, D=1024, fp32 in/out.
#define LSEQ   1024
#define DDIM   1024
#define NBATCH 16
#define S_LD   1028   // padded fp32 stride for S rows -> conflict-free frag reads
#define VT_LD  40     // padded f16 stride (fallback path LDS transpose)
#define TT_LD  72     // padded f16 stride for 64x64 transpose tiles

typedef __attribute__((ext_vector_type(16))) _Float16 v16h;
typedef __attribute__((ext_vector_type(8)))  _Float16 v8h;
typedef __attribute__((ext_vector_type(4)))  _Float16 v4h;
typedef __attribute__((ext_vector_type(8)))  float    v8f;
typedef __attribute__((ext_vector_type(4)))  float    v4f;

__device__ __forceinline__ v8f zero8() {
  v8f z;
#pragma unroll
  for (int i = 0; i < 8; ++i) z[i] = 0.f;
  return z;
}

__device__ __forceinline__ v16h pack_runs(v4f r0, v4f r1, v4f r2, v4f r3) {
  v16h h;
#pragma unroll
  for (int i = 0; i < 4; ++i) {
    h[i]      = (_Float16)r0[i];
    h[i + 4]  = (_Float16)r1[i];
    h[i + 8]  = (_Float16)r2[i];
    h[i + 12] = (_Float16)r3[i];
  }
  return h;
}

__device__ __forceinline__ v16h cat8(v8h lo, v8h hi) {
  return __builtin_shufflevector(lo, hi, 0, 1, 2, 3, 4, 5, 6, 7,
                                 8, 9, 10, 11, 12, 13, 14, 15);
}

// --- prep: fp32 -> f16 copy + inverse L2 norm, one wave per row -----------
__global__ __launch_bounds__(256)
void prep_kernel(const float* __restrict__ X, _Float16* __restrict__ X16,
                 float* __restrict__ inv) {
  const int row  = blockIdx.x * 8 + ((int)threadIdx.x >> 5);
  const int lane = (int)threadIdx.x & 31;
  const float* x = X + (size_t)row * DDIM;
  _Float16*    y = X16 + (size_t)row * DDIM;
  float s = 0.f;
#pragma unroll
  for (int it = 0; it < DDIM / 128; ++it) {
    v4f v = *(const v4f*)(x + it * 128 + lane * 4);
    s += v[0] * v[0] + v[1] * v[1] + v[2] * v[2] + v[3] * v[3];
    v4h h;
#pragma unroll
    for (int c = 0; c < 4; ++c) h[c] = (_Float16)v[c];
    *(v4h*)(y + it * 128 + lane * 4) = h;
  }
#pragma unroll
  for (int off = 16; off > 0; off >>= 1) s += __shfl_down(s, off, 32);
  if (lane == 0) inv[row] = rsqrtf(s);
}

// --- one-time f16 transpose: XT[b][d][j] = X[b][j][d], 64x64 LDS tiles ----
__global__ __launch_bounds__(256)
void transpose16_kernel(const _Float16* __restrict__ X,
                        _Float16* __restrict__ XT) {
  __shared__ _Float16 T[64 * TT_LD];
  const int b  = blockIdx.z;
  const int tj = blockIdx.y;   // 64-row tile in j
  const int td = blockIdx.x;   // 64-col tile in d
  const int t  = (int)threadIdx.x;

  const _Float16* Xb  = X + (size_t)b * LSEQ * DDIM;
  _Float16*       XTb = XT + (size_t)b * DDIM * LSEQ;

  {  // load 4x4 micro-tiles, transpose in registers, store to LDS
    const int rg = t >> 4, cg = t & 15;
    const int j = tj * 64 + rg * 4;
    const int d = td * 64 + cg * 4;
    const _Float16* src = Xb + (size_t)j * DDIM + d;
    v4h x0 = *(const v4h*)(src);
    v4h x1 = *(const v4h*)(src + DDIM);
    v4h x2 = *(const v4h*)(src + 2 * DDIM);
    v4h x3 = *(const v4h*)(src + 3 * DDIM);
#pragma unroll
    for (int c = 0; c < 4; ++c) {
      v4h w;
      w[0] = x0[c]; w[1] = x1[c]; w[2] = x2[c]; w[3] = x3[c];
      *(v4h*)&T[(cg * 4 + c) * TT_LD + rg * 4] = w;
    }
  }
  __syncthreads();
  {  // coalesced writes of transposed rows
    const int dloc = t >> 2, seg = t & 3;
    v8h r0 = *(const v8h*)&T[dloc * TT_LD + seg * 16];
    v8h r1 = *(const v8h*)&T[dloc * TT_LD + seg * 16 + 8];
    _Float16* orow = XTb + (size_t)(td * 64 + dloc) * LSEQ + tj * 64 + seg * 16;
    *(v8h*)(orow)     = r0;
    *(v8h*)(orow + 8) = r1;
  }
}

// --- norms only (fallback path) -------------------------------------------
__global__ __launch_bounds__(256)
void rownorm_kernel(const float* __restrict__ X, float* __restrict__ inv) {
  const int row  = blockIdx.x * 8 + ((int)threadIdx.x >> 5);
  const int lane = (int)threadIdx.x & 31;
  const float* x = X + (size_t)row * DDIM;
  float s = 0.f;
#pragma unroll
  for (int it = 0; it < DDIM / 128; ++it) {
    v4f v = *(const v4f*)(x + it * 128 + lane * 4);
    s += v[0] * v[0] + v[1] * v[1] + v[2] * v[2] + v[3] * v[3];
  }
#pragma unroll
  for (int off = 16; off > 0; off >>= 1) s += __shfl_down(s, off, 32);
  if (lane == 0) inv[row] = rsqrtf(s);
}

// ===========================================================================
// Primary path: f16 operands + pre-transposed f16 V in workspace.
// Both phases are pure global_load_b128 + WMMA streams (no staging).
// Out = softmax((Q Kt) * invq*invk) @ K ; one WG (8 waves) per 16-query tile.
// ===========================================================================
__global__ __launch_bounds__(256)
void attend_full_kernel(const _Float16* __restrict__ Q16,
                        const _Float16* __restrict__ K16,   // [b][j][d]
                        const _Float16* __restrict__ KT16,  // [b][d][j]
                        const float* __restrict__ invq,
                        const float* __restrict__ invk,
                        float* __restrict__ Out) {
  __shared__ float S[16 * S_LD];   // logits -> exp weights
  __shared__ float iq[16];
  __shared__ float red[256];
  __shared__ float rinv[16];

  const int b     = blockIdx.y;
  const int qbase = blockIdx.x << 4;
  const int tid   = (int)threadIdx.x;
  const int wave  = tid >> 5;
  const int lane  = tid & 31;
  const int half  = (lane >> 4) & 1;
  const int l15   = lane & 15;
  const int wnb   = wave << 7;     // this wave owns 128 N-columns

  const _Float16* Qb  = Q16 + ((size_t)b * LSEQ + qbase) * DDIM;
  const _Float16* Kb  = K16 + (size_t)b * LSEQ * DDIM;
  const _Float16* KTb = KT16 + (size_t)b * DDIM * LSEQ;
  float*          Ob  = Out + ((size_t)b * LSEQ + qbase) * DDIM;

  if (tid < 16) iq[tid] = invq[b * LSEQ + qbase + tid];
  __syncthreads();

  // ---------------- Phase 1: S[16 x 1024] = Qn @ Knt ----------------------
  {
    v8f acc[8];
#pragma unroll
    for (int t = 0; t < 8; ++t) acc[t] = zero8();

    const _Float16* arow = Qb + (size_t)l15 * DDIM + (half << 3);
    for (int kc = 0; kc < DDIM / 32; ++kc) {
      const int k0 = kc << 5;
      v8h a_lo = *(const v8h*)(arow + k0);
      v8h a_hi = *(const v8h*)(arow + k0 + 16);
      v16h afrag = cat8(a_lo, a_hi);
#pragma unroll
      for (int nt = 0; nt < 8; ++nt) {
        const int n = wnb + (nt << 4) + l15;
        const _Float16* brow = Kb + (size_t)n * DDIM + k0 + (half << 3);
        v16h bfrag = cat8(*(const v8h*)(brow), *(const v8h*)(brow + 16));
        acc[nt] = __builtin_amdgcn_wmma_f32_16x16x32_f16(
            false, afrag, false, bfrag, (short)0, acc[nt], false, false);
      }
    }
#pragma unroll
    for (int nt = 0; nt < 8; ++nt) {
      const int n = wnb + (nt << 4) + l15;
      const float vk = invk[b * LSEQ + n];
#pragma unroll
      for (int v = 0; v < 8; ++v) {
        const int m = v + (half << 3);
        S[m * S_LD + n] = acc[nt][v] * iq[m] * vk;
      }
    }
  }
  __syncthreads();

  // ------------- softmax: logits are cosines in [-1,1] -------------------
  {
    const int r = tid >> 4, c = tid & 15;
    float part = 0.f;
    for (int j = c; j < LSEQ; j += 16) {
      float e = __expf(S[r * S_LD + j]);
      S[r * S_LD + j] = e;
      part += e;
    }
    red[tid] = part;
  }
  __syncthreads();
  if (tid < 16) {
    float s = 0.f;
#pragma unroll
    for (int c = 0; c < 16; ++c) s += red[tid * 16 + c];
    rinv[tid] = 1.0f / s;
  }
  __syncthreads();

  // -------- Phase 2: Out[16 x 1024] = P @ V, V read via transposed copy ---
  {
    v8f acc[8];
#pragma unroll
    for (int t = 0; t < 8; ++t) acc[t] = zero8();

    for (int kc = 0; kc < LSEQ / 32; ++kc) {
      const int k0 = kc << 5;
      // A fragment: exp weights from S (fp32 -> f16, 8 cvts per kc)
      const float* prow = &S[l15 * S_LD + k0 + (half << 3)];
      v16h afrag = pack_runs(*(const v4f*)(prow), *(const v4f*)(prow + 4),
                             *(const v4f*)(prow + 16), *(const v4f*)(prow + 20));
#pragma unroll
      for (int nt = 0; nt < 8; ++nt) {
        const int n = wnb + (nt << 4) + l15;   // output d-column
        const _Float16* brow = KTb + (size_t)n * LSEQ + k0 + (half << 3);
        v16h bfrag = cat8(*(const v8h*)(brow), *(const v8h*)(brow + 16));
        acc[nt] = __builtin_amdgcn_wmma_f32_16x16x32_f16(
            false, afrag, false, bfrag, (short)0, acc[nt], false, false);
      }
    }

#pragma unroll
    for (int nt = 0; nt < 8; ++nt) {
      const int n = wnb + (nt << 4) + l15;
#pragma unroll
      for (int v = 0; v < 8; ++v) {
        const int m = v + (half << 3);
        Ob[(size_t)m * DDIM + n] = acc[nt][v] * rinv[m];
      }
    }
  }
}

// ===========================================================================
// Fallback path: fp32 operands converted on the fly, LDS V transpose.
// ===========================================================================
__global__ __launch_bounds__(256)
void attend_kernel(const float* __restrict__ Q, const float* __restrict__ K,
                   const float* __restrict__ invq, const float* __restrict__ invk,
                   float* __restrict__ Out) {
  __shared__ float    S[16 * S_LD];
  __shared__ _Float16 Vt[LSEQ * VT_LD];
  __shared__ float    iq[16];
  __shared__ float    red[256];
  __shared__ float    rinv[16];

  const int b     = blockIdx.y;
  const int qbase = blockIdx.x << 4;
  const int tid   = (int)threadIdx.x;
  const int wave  = tid >> 5;
  const int lane  = tid & 31;
  const int half  = (lane >> 4) & 1;
  const int l15   = lane & 15;
  const int wnb   = wave << 7;

  const float* Qb = Q + ((size_t)b * LSEQ + qbase) * DDIM;
  const float* Kb = K + (size_t)b * LSEQ * DDIM;
  float*       Ob = Out + ((size_t)b * LSEQ + qbase) * DDIM;

  if (tid < 16) iq[tid] = invq[b * LSEQ + qbase + tid];
  __syncthreads();

  {
    v8f acc[8];
#pragma unroll
    for (int t = 0; t < 8; ++t) acc[t] = zero8();

    const float* arow = Qb + (size_t)l15 * DDIM + (half << 3);
    for (int kc = 0; kc < DDIM / 32; ++kc) {
      const int k0 = kc << 5;
      v16h afrag = pack_runs(*(const v4f*)(arow + k0), *(const v4f*)(arow + k0 + 4),
                             *(const v4f*)(arow + k0 + 16), *(const v4f*)(arow + k0 + 20));
#pragma unroll
      for (int nt = 0; nt < 8; ++nt) {
        const int n = wnb + (nt << 4) + l15;
        const float* brow = Kb + (size_t)n * DDIM + k0 + (half << 3);
        v16h bfrag = pack_runs(*(const v4f*)(brow), *(const v4f*)(brow + 4),
                               *(const v4f*)(brow + 16), *(const v4f*)(brow + 20));
        acc[nt] = __builtin_amdgcn_wmma_f32_16x16x32_f16(
            false, afrag, false, bfrag, (short)0, acc[nt], false, false);
      }
    }
#pragma unroll
    for (int nt = 0; nt < 8; ++nt) {
      const int n = wnb + (nt << 4) + l15;
      const float vk = invk[b * LSEQ + n];
#pragma unroll
      for (int v = 0; v < 8; ++v) {
        const int m = v + (half << 3);
        S[m * S_LD + n] = acc[nt][v] * iq[m] * vk;
      }
    }
  }
  __syncthreads();

  {
    const int r = tid >> 4, c = tid & 15;
    float part = 0.f;
    for (int j = c; j < LSEQ; j += 16) {
      float e = __expf(S[r * S_LD + j]);
      S[r * S_LD + j] = e;
      part += e;
    }
    red[tid] = part;
  }
  __syncthreads();
  if (tid < 16) {
    float s = 0.f;
#pragma unroll
    for (int c = 0; c < 16; ++c) s += red[tid * 16 + c];
    rinv[tid] = 1.0f / s;
  }

  {
    v8f acc[8];
#pragma unroll
    for (int t = 0; t < 8; ++t) acc[t] = zero8();

    const int jg = (tid >> 5) << 2;
    const int cb = (tid & 31) << 2;

    for (int kc = 0; kc < LSEQ / 32; ++kc) {
      __syncthreads();
      const float* vbase = Kb + (size_t)(kc << 5) * DDIM;
#pragma unroll
      for (int p = 0; p < 8; ++p) {
        const int n0 = (p << 7) + cb;
        const float* r0 = vbase + (size_t)jg * DDIM + n0;
        v4f x0 = *(const v4f*)(r0);
        v4f x1 = *(const v4f*)(r0 + DDIM);
        v4f x2 = *(const v4f*)(r0 + 2 * DDIM);
        v4f x3 = *(const v4f*)(r0 + 3 * DDIM);
#pragma unroll
        for (int c = 0; c < 4; ++c) {
          v4h t;
          t[0] = (_Float16)x0[c]; t[1] = (_Float16)x1[c];
          t[2] = (_Float16)x2[c]; t[3] = (_Float16)x3[c];
          *(v4h*)&Vt[(size_t)(n0 + c) * VT_LD + jg] = t;
        }
      }
      __syncthreads();

      const int k0 = kc << 5;
      const float* prow = &S[l15 * S_LD + k0 + (half << 3)];
      v16h afrag = pack_runs(*(const v4f*)(prow), *(const v4f*)(prow + 4),
                             *(const v4f*)(prow + 16), *(const v4f*)(prow + 20));
#pragma unroll
      for (int nt = 0; nt < 8; ++nt) {
        const int n = wnb + (nt << 4) + l15;
        const v8h lo = *(const v8h*)&Vt[(size_t)n * VT_LD + (half << 3)];
        const v8h hi = *(const v8h*)&Vt[(size_t)n * VT_LD + (half << 3) + 16];
        v16h bfrag = cat8(lo, hi);
        acc[nt] = __builtin_amdgcn_wmma_f32_16x16x32_f16(
            false, afrag, false, bfrag, (short)0, acc[nt], false, false);
      }
    }

#pragma unroll
    for (int nt = 0; nt < 8; ++nt) {
      const int n = wnb + (nt << 4) + l15;
#pragma unroll
      for (int v = 0; v < 8; ++v) {
        const int m = v + (half << 3);
        Ob[(size_t)m * DDIM + n] = acc[nt][v] * rinv[m];
      }
    }
  }
}

extern "C" void kernel_launch(void* const* d_in, const int* in_sizes, int n_in,
                              void* d_out, int out_size, void* d_ws, size_t ws_size,
                              hipStream_t stream) {
  (void)in_sizes; (void)n_in; (void)out_size;
  const float* p = (const float*)d_in[0];  // premise    [16,1024,1024] f32
  const float* h = (const float*)d_in[1];  // hypothesis [16,1024,1024] f32

  float* out_a = (float*)d_out;                        // a_tilde
  float* out_b = out_a + (size_t)NBATCH * LSEQ * DDIM; // b_tilde

  const size_t nrows = (size_t)NBATCH * LSEQ;
  const size_t nelem = nrows * DDIM;
  // inv norms (2x) + f16 copies (2x) + transposed f16 copies (2x)
  const size_t need  = 2 * nrows * sizeof(float) + 4 * nelem * sizeof(_Float16);

  dim3 grid(LSEQ / 16, NBATCH);

  if (ws_size >= need) {
    // ---- primary path ----
    float*    invp = (float*)d_ws;
    float*    invh = invp + nrows;
    _Float16* p16  = (_Float16*)(invh + nrows);
    _Float16* h16  = p16 + nelem;
    _Float16* pT16 = h16 + nelem;
    _Float16* hT16 = pT16 + nelem;

    prep_kernel<<<(int)(nrows / 8), 256, 0, stream>>>(p, p16, invp);
    prep_kernel<<<(int)(nrows / 8), 256, 0, stream>>>(h, h16, invh);

    dim3 tgrid(DDIM / 64, LSEQ / 64, NBATCH);
    transpose16_kernel<<<tgrid, 256, 0, stream>>>(p16, pT16);
    transpose16_kernel<<<tgrid, 256, 0, stream>>>(h16, hT16);

    // a_tilde: Q=premise, K=V=hypothesis
    attend_full_kernel<<<grid, 256, 0, stream>>>(p16, h16, hT16, invp, invh, out_a);
    // b_tilde: Q=hypothesis, K=V=premise
    attend_full_kernel<<<grid, 256, 0, stream>>>(h16, p16, pT16, invh, invp, out_b);
  } else {
    // ---- fallback: only norms in workspace (128 KB) ----
    float* invp = (float*)d_ws;
    float* invh = invp + nrows;

    rownorm_kernel<<<(int)(nrows / 8), 256, 0, stream>>>(p, invp);
    rownorm_kernel<<<(int)(nrows / 8), 256, 0, stream>>>(h, invh);

    attend_kernel<<<grid, 256, 0, stream>>>(p, h, invp, invh, out_a);
    attend_kernel<<<grid, 256, 0, stream>>>(h, p, invh, invp, out_b);
  }
}
